// ClauseRec_86165633892476
// MI455X (gfx1250) — matrix operand order repaired and checked by reading, not compiled, not run on hardware
//
#include <hip/hip_runtime.h>

// ---------------------------------------------------------------------------
// CDNA5 (gfx1250) GNN forward: 2x SAGEConv + GraphConv + linear/softmax(dim=1)
//   N nodes, E edges, feature dim 64 throughout.
//   Dense matmuls via v_wmma_f32_16x16x32_f16 (wave32, fp32 accumulate).
//   Weight staging into padded LDS via TDM tensor_load_to_lds (TENSORcnt).
//   Aggregation via fp32 global atomics (all arrays are L2-resident: 12.8 MB).
//   softmax over a size-1 axis == 1.0 exactly, so d_out = ones(N).
// ---------------------------------------------------------------------------

typedef __attribute__((ext_vector_type(16))) _Float16 v16h;
typedef __attribute__((ext_vector_type(8)))  float    v8f;
typedef __attribute__((ext_vector_type(4)))  unsigned int u32x4;
typedef __attribute__((ext_vector_type(8)))  unsigned int u32x8;
typedef unsigned int uint32;

#define STRH 68   // padded LDS row stride in halfs (64 + 4) to avoid bank conflicts

// ---------------------------- tiny utility kernels -------------------------

__global__ void zero_f32_kernel(float* __restrict__ p, int count) {
  int i = blockIdx.x * blockDim.x + threadIdx.x;
  if (i < count) p[i] = 0.0f;
}

__global__ void ones_f32_kernel(float* __restrict__ p, int count) {
  int i = blockIdx.x * blockDim.x + threadIdx.x;
  if (i < count) p[i] = 1.0f;   // softmax over a singleton axis is exactly 1
}

__global__ void cvt_f32_to_f16_kernel(const float* __restrict__ src,
                                      _Float16* __restrict__ dst, int count) {
  int i = blockIdx.x * blockDim.x + threadIdx.x;
  if (i < count) dst[i] = (_Float16)src[i];
}

__global__ void degree_kernel(const int* __restrict__ dst_idx,
                              float* __restrict__ deg, int E) {
  int e = blockIdx.x * blockDim.x + threadIdx.x;
  if (e < E) atomicAdd(&deg[dst_idx[e]], 1.0f);
}

// 16 threads per edge; each does a float4 gather + 4 fp32 atomic adds.
// Feature matrices are 12.8 MB -> resident in the 192 MB L2.
__global__ void scatter_add_kernel(const float* __restrict__ feat,
                                   const int* __restrict__ src_idx,
                                   const int* __restrict__ dst_idx,
                                   float* __restrict__ agg, int E) {
  int tid = blockIdx.x * blockDim.x + threadIdx.x;
  int total = E * 16;
  if (tid >= total) return;
  int e = tid >> 4;
  int q = tid & 15;
  int s = src_idx[e];
  int d = dst_idx[e];
  const float4 v = *(const float4*)(feat + (size_t)s * 64 + q * 4);
  float* p = agg + (size_t)d * 64 + q * 4;
  atomicAdd(p + 0, v.x);
  atomicAdd(p + 1, v.y);
  atomicAdd(p + 2, v.z);
  atomicAdd(p + 3, v.w);
}

// ----------------------- Tensor Data Mover (TDM) ---------------------------
// DMA one 64x64 f16 weight matrix (row-major, contiguous) from global memory
// into LDS at lds_off, inserting 2 DWORDs (4 halfs) of padding after every
// 32 DWORDs (one 64-half row) -> padded LDS row stride of 68 halfs.
// D# encoding per CDNA5 ISA (2-D tensor: groups 0+1).
__device__ __forceinline__ void tdm_load_weight_64x64(const _Float16* gsrc,
                                                      uint32 lds_off) {
  unsigned long long ga = (unsigned long long)(size_t)gsrc;
  u32x4 g0;
  g0[0] = 1u;                                  // count=1 (valid), user mode
  g0[1] = lds_off;                             // lds_addr (bytes)
  g0[2] = (uint32)ga;                          // global_addr[31:0]
  g0[3] = (uint32)((ga >> 32) & 0x01FFFFFFull) // global_addr[56:32]
          | (2u << 30);                        // type=2 ("image")
  u32x8 g1;
  g1[0] = (1u << 16)                           // data_size = 1 -> 2 bytes
        | (1u << 20)                           // pad_enable
        | (4u << 22)                           // pad_interval: 2<<4 = 32 DWORDs
        | (1u << 25);                          // pad_amount: 2 DWORDs
  g1[1] = (64u << 16);                         // tensor_dim0 = 64
  g1[2] = (64u << 16);                         // tensor_dim1 = 64
  g1[3] = (64u << 16);                         // tile_dim0   = 64
  g1[4] = 64u;                                 // tile_dim1=64, tile_dim2=0
  g1[5] = 64u;                                 // tensor_dim0_stride = 64
  g1[6] = 0u;
  g1[7] = 0u;
  // 2-D tensor: VADDR2/VADDR3 omitted (NULL). Tracked by TENSORcnt.
  asm volatile("tensor_load_to_lds %0, %1" :: "s"(g0), "s"(g1) : "memory");
}

// ----------------------- WMMA fragment construction ------------------------
// A-matrix 16x32 f16 (ISA 7.12.2): lane<16 -> M=lane, VGPR p in [0,3]: K=2p,2p+1,
//   VGPR p in [4,7]: K=16+2(p-4)+{0,1}; lanes 16-31 same with K offset +8.
//   => K(p,h) = 32*s + 8*h + 2p + (p>=4 ? 8 : 0)
__device__ __forceinline__ v16h frag_a(const _Float16* __restrict__ row, int h, int s) {
  union { v16h v; uint32 u[8]; } f;
#pragma unroll
  for (int p = 0; p < 8; ++p) {
    int K = 32 * s + 8 * h + 2 * p + ((p >= 4) ? 8 : 0);
    f.u[p] = *(const uint32*)(row + K);
  }
  return f.v;
}

// B-matrix 32x16 f16: lane%16 = N; same K order as A fragment.
//   => K(p,h) = 32*s + 8*h + 2p + (p>=4 ? 8 : 0)
__device__ __forceinline__ v16h frag_b(const _Float16* __restrict__ row, int h, int s) {
  union { v16h v; uint32 u[8]; } f;
#pragma unroll
  for (int p = 0; p < 8; ++p) {
    int K = 32 * s + 8 * h + 2 * p + ((p >= 4) ? 8 : 0);
    f.u[p] = *(const uint32*)(row + K);
  }
  return f.v;
}

// ------------------------------ dense layer --------------------------------
// out[n,:] = relu( (agg[n,:] * scale_n) @ Wl^T + xin[n,:] @ Wr^T + bias )
//   scale_n = 1/max(deg[n],1) for mean aggregation, 1 for sum aggregation.
// One wave per 16-row tile; 8 waves (256 threads) per block.
// 16 x v_wmma_f32_16x16x32_f16 per wave (4 column tiles x 2 matmuls x K=64).
__global__ __launch_bounds__(256)
void gnn_dense_wmma(const float* __restrict__ xin,
                    const float* __restrict__ agg,
                    const float* __restrict__ deg,
                    const _Float16* __restrict__ Wl,   // [64][64] row-major, Wl[o][k]
                    const _Float16* __restrict__ Wr,   // [64][64]
                    const float* __restrict__ bias,    // [64]
                    float* __restrict__ out,           // [n][64]
                    int n, int use_mean) {
  __shared__ _Float16 sW[2][64 * STRH];   // weights (block-shared, TDM-filled)
  __shared__ _Float16 sA[8][16 * STRH];   // scaled agg rows (per wave)
  __shared__ _Float16 sX[8][16 * STRH];   // input rows      (per wave)

  const int tid  = threadIdx.x;
  const int lane = tid & 31;
  // Force wave/tile scalar (SGPR) so tile-level branches are SALU-uniform.
  const int wave = __builtin_amdgcn_readfirstlane(tid >> 5);

  // ---- stage both weight matrices to LDS via the Tensor Data Mover ----
  // Issue from the first wave only (scalar-gated); EXEC is ignored by TDM.
  if (__builtin_amdgcn_readfirstlane(tid) == 0) {
    uint32 sw0 = (uint32)(size_t)(&sW[0][0]);          // LDS byte offset
    tdm_load_weight_64x64(Wl, sw0);
    tdm_load_weight_64x64(Wr, sw0 + 64u * STRH * 2u);
    __builtin_amdgcn_s_wait_tensorcnt(0);
  }

  const int ntiles = (n + 15) >> 4;
  const int tile   = blockIdx.x * 8 + wave;

  // ---- stage this wave's 16 node rows (coalesced float4, convert to f16) ----
  if (tile < ntiles) {
    const int row_base = tile * 16;
    for (int i = lane; i < 256; i += 32) {     // 16 rows x 16 float4
      int r    = i >> 4;
      int c4   = i & 15;
      int node = row_base + r;
      if (node >= n) node = n - 1;             // clamp: keeps EXEC all-ones later
      float4 va = *(const float4*)(agg + (size_t)node * 64 + c4 * 4);
      float4 vx = *(const float4*)(xin + (size_t)node * 64 + c4 * 4);
      float sc = 1.0f;
      if (use_mean) sc = 1.0f / fmaxf(deg[node], 1.0f);
      _Float16* pa = &sA[wave][r * STRH + c4 * 4];
      _Float16* px = &sX[wave][r * STRH + c4 * 4];
      pa[0] = (_Float16)(va.x * sc); pa[1] = (_Float16)(va.y * sc);
      pa[2] = (_Float16)(va.z * sc); pa[3] = (_Float16)(va.w * sc);
      px[0] = (_Float16)vx.x; px[1] = (_Float16)vx.y;
      px[2] = (_Float16)vx.z; px[3] = (_Float16)vx.w;
    }
  }
  __syncthreads();
  if (tile >= ntiles) return;                  // uniform (scalar) wave exit

  const int m = lane & 15;                     // row within tile (A) / col (B,D)
  const int h = lane >> 4;                     // half-wave select
  const _Float16* rowA = &sA[wave][m * STRH];
  const _Float16* rowX = &sX[wave][m * STRH];

  const v16h aL0 = frag_a(rowA, h, 0), aL1 = frag_a(rowA, h, 1);
  const v16h aR0 = frag_a(rowX, h, 0), aR1 = frag_a(rowX, h, 1);

  const int row_base = tile * 16;
  const bool full_tile = (row_base + 16 <= n); // uniform: fast path, no guards

#pragma unroll
  for (int nt = 0; nt < 4; ++nt) {
    const int o = nt * 16 + m;                 // output channel for this lane
    const _Float16* rowWl = &sW[0][o * STRH];
    const _Float16* rowWr = &sW[1][o * STRH];
    v8f c = {};
    c = __builtin_amdgcn_wmma_f32_16x16x32_f16(false, aL0, false, frag_b(rowWl, h, 0),
                                               (short)0, c, false, false);
    c = __builtin_amdgcn_wmma_f32_16x16x32_f16(false, aL1, false, frag_b(rowWl, h, 1),
                                               (short)0, c, false, false);
    c = __builtin_amdgcn_wmma_f32_16x16x32_f16(false, aR0, false, frag_b(rowWr, h, 0),
                                               (short)0, c, false, false);
    c = __builtin_amdgcn_wmma_f32_16x16x32_f16(false, aR1, false, frag_b(rowWr, h, 1),
                                               (short)0, c, false, false);
    const float bb = bias[o];
    float* colp = out + (size_t)(row_base + 8 * h) * 64 + o;
    if (full_tile) {                           // D: M = r + 8*h, N = o
#pragma unroll
      for (int r = 0; r < 8; ++r)
        colp[(size_t)r * 64] = fmaxf(c[r] + bb, 0.0f);
    } else {
#pragma unroll
      for (int r = 0; r < 8; ++r) {
        int node = row_base + r + 8 * h;
        if (node < n) out[(size_t)node * 64 + o] = fmaxf(c[r] + bb, 0.0f);
      }
    }
  }
}

// ------------------------------- launcher ----------------------------------
// Inputs (setup_inputs order):
//  0:x[N*64] 1:Wl1 2:Wr1 3:b1 4:Wl2 5:Wr2 6:b2 7:Wrel3 8:Wroot3 9:b3
//  10:Wlin 11:blin 12:edge_index[2*E] (int32 per harness convention)
extern "C" void kernel_launch(void* const* d_in, const int* in_sizes, int n_in,
                              void* d_out, int out_size, void* d_ws, size_t ws_size,
                              hipStream_t stream) {
  const int N = in_sizes[0] / 64;
  const int E = in_sizes[12] / 2;

  const float* x    = (const float*)d_in[0];
  const float* b1   = (const float*)d_in[3];
  const float* b2   = (const float*)d_in[6];
  const float* b3   = (const float*)d_in[9];
  const int*   edge = (const int*)d_in[12];
  const int*   src  = edge;
  const int*   dst  = edge + E;

  // ---- workspace carve-up (256B aligned) ----
  char* ws = (char*)d_ws;
  size_t off = 0;
  auto carve = [&](size_t bytes) -> char* {
    char* p = ws + off;
    off = (off + bytes + 255) & ~(size_t)255;
    return p;
  };
  _Float16* w16 = (_Float16*)carve(6 * 64 * 64 * sizeof(_Float16)); // f16 weights
  float* deg = (float*)carve((size_t)N * sizeof(float));
  float* agg = (float*)carve((size_t)N * 64 * sizeof(float));
  float* h1  = (float*)carve((size_t)N * 64 * sizeof(float));
  float* h2  = (float*)carve((size_t)N * 64 * sizeof(float));
  (void)ws_size; (void)n_in;

  const int NB    = 256;
  const int gF    = (N * 64 + NB - 1) / NB;     // feature-matrix sized grids
  const int gN    = (N + NB - 1) / NB;
  const int gE    = (E + NB - 1) / NB;
  const int gS    = (E * 16 + NB - 1) / NB;     // scatter grid
  const int tiles = (N + 15) / 16;
  const int gD    = (tiles + 7) / 8;            // dense grid: 8 waves/block

  // ---- one-time per launch: f16 weights + degrees ----
  const int widx[6] = {1, 2, 4, 5, 7, 8};       // Wl1 Wr1 Wl2 Wr2 Wrel3 Wroot3
  for (int k = 0; k < 6; ++k) {
    cvt_f32_to_f16_kernel<<<(4096 + NB - 1) / NB, NB, 0, stream>>>(
        (const float*)d_in[widx[k]], w16 + (size_t)k * 4096, 4096);
  }
  zero_f32_kernel<<<gN, NB, 0, stream>>>(deg, N);
  degree_kernel<<<gE, NB, 0, stream>>>(dst, deg, E);

  // ---- layer 1: SAGEConv(x) -> h1 (mean aggregation) ----
  zero_f32_kernel<<<gF, NB, 0, stream>>>(agg, N * 64);
  scatter_add_kernel<<<gS, NB, 0, stream>>>(x, src, dst, agg, E);
  gnn_dense_wmma<<<gD, NB, 0, stream>>>(x, agg, deg, w16 + 0 * 4096, w16 + 1 * 4096,
                                        b1, h1, N, 1);

  // ---- layer 2: SAGEConv(h1) -> h2 (mean aggregation) ----
  zero_f32_kernel<<<gF, NB, 0, stream>>>(agg, N * 64);
  scatter_add_kernel<<<gS, NB, 0, stream>>>(h1, src, dst, agg, E);
  gnn_dense_wmma<<<gD, NB, 0, stream>>>(h1, agg, deg, w16 + 2 * 4096, w16 + 3 * 4096,
                                        b2, h2, N, 1);

  // ---- layer 3: GraphConv(h2) -> h1 (sum aggregation; buffer reuse) ----
  zero_f32_kernel<<<gF, NB, 0, stream>>>(agg, N * 64);
  scatter_add_kernel<<<gS, NB, 0, stream>>>(h2, src, dst, agg, E);
  gnn_dense_wmma<<<gD, NB, 0, stream>>>(h2, agg, deg, w16 + 4 * 4096, w16 + 5 * 4096,
                                        b3, h1, N, 0);

  // ---- head: softmax over a size-1 axis is identically 1.0 ----
  ones_f32_kernel<<<(out_size + NB - 1) / NB, NB, 0, stream>>>((float*)d_out, out_size);
}
